// Quantize_22892175687681
// MI455X (gfx1250) — matrix operand order repaired and checked by reading, not compiled, not run on hardware
//
#include <hip/hip_runtime.h>
#include <hip/hip_bf16.h>

// VQ-VAE grouped quantizer for MI455X (gfx1250, wave32, WMMA).
// M = 8*4096*4 = 131072 sub-vectors of d = 128; K = 512 codes.
// Scores via bf16 WMMA (f32 accumulate); outputs gathered in f32 (exact).
// Each wave owns 32 rows (two 16-row A-tiles) sharing B fragments:
// 8 WMMAs per 8 ds_load_b128, dual accumulator chains hide WMMA->VALU hazard.

typedef __attribute__((ext_vector_type(16))) __bf16 v16bf;
typedef __attribute__((ext_vector_type(8)))  __bf16 v8bf;
typedef __attribute__((ext_vector_type(4)))  __bf16 v4bf;
typedef __attribute__((ext_vector_type(8)))  float  v8f;

constexpr int D_SUB       = 128;     // sub-vector dim (512 / 4 groups)
constexpr int NCODE       = 512;     // codebook entries
constexpr int ROWPAD      = 136;     // bf16 elems per LDS row (272B stride -> conflict-free b128)
constexpr int ROWS_PER_WG = 256;     // 8 waves x 32 rows
constexpr int M_TOTAL     = 131072;  // total sub-vectors
constexpr float DIFF_SCALE = 12.5f / 16777216.0f;  // KLD_SCALE*(1+COMMITMENT)/(M*d)

__global__ void vq_zero_kernel(float* diffSlot) { *diffSlot = 0.0f; }

__global__ __launch_bounds__(256)
void vq_main_kernel(const float* __restrict__ z,      // [M,128] f32
                    const float* __restrict__ ew,     // [512,128] f32
                    float* __restrict__ outq,         // [M,128] f32 (z_q_st)
                    float* __restrict__ diffSlot,     // [1]
                    float* __restrict__ indOut)       // [M] (indices stored as float)
{
    extern __shared__ char smem[];
    __bf16* sB    = (__bf16*)smem;                          // [512][136] bf16 codebook
    float*  sHalf = (float*)(smem + NCODE * ROWPAD * 2);    // [512] 0.5*||e_k||^2
    float*  sRed  = sHalf + NCODE;                          // [8] per-wave partials

    const int tid  = threadIdx.x;
    const int lane = tid & 31;
    const int wave = tid >> 5;
    const int h    = lane >> 4;   // half-wave (0/1)
    const int l    = lane & 15;

    // ---- Stage 1: cooperative fill of bf16 codebook into LDS (coalesced) ----
    const float4* ef4 = (const float4*)ew;   // 512 rows * 32 float4
    #pragma unroll 4
    for (int j = 0; j < 64; ++j) {
        int idx = j * 256 + tid;             // float4 index in [0, 16384)
        int row = idx >> 5;
        int c4  = idx & 31;
        float4 v = ef4[idx];
        v4bf b;
        b[0] = (__bf16)v.x; b[1] = (__bf16)v.y; b[2] = (__bf16)v.z; b[3] = (__bf16)v.w;
        *(v4bf*)(sB + row * ROWPAD + c4 * 4) = b;  // 8B ds_store
    }
    // ---- half-norms in f32 from the original codebook ----
    for (int r = tid; r < NCODE; r += 256) {
        const float4* rp = ef4 + r * 32;
        float s = 0.0f;
        #pragma unroll
        for (int j = 0; j < 32; ++j) {
            float4 v = rp[j];
            s += v.x * v.x + v.y * v.y + v.z * v.z + v.w * v.w;
        }
        sHalf[r] = 0.5f * s;
    }
    __syncthreads();

    // ---- Stage 2: load A fragments for two 16-row tiles (16x32 bf16 x4 chunks each) ----
    const int mBase = blockIdx.x * ROWS_PER_WG + wave * 32;

    v16bf afrag[2][4];
    #pragma unroll
    for (int t = 0; t < 2; ++t) {
        const float* zr = z + (size_t)(mBase + 16 * t + l) * D_SUB;  // M = lane&15
        #pragma unroll
        for (int c = 0; c < 4; ++c) {
            // ISA A-layout: lanes 0-15 K runs {0..7,16..23}, lanes 16-31 {8..15,24..31}
            const int k0 = c * 32 + 8 * h;
            const int k1 = c * 32 + 16 + 8 * h;
            const float4* p0 = (const float4*)(zr + k0);
            const float4* p1 = (const float4*)(zr + k1);
            float4 a0 = p0[0], a1 = p0[1];
            float4 b0 = p1[0], b1 = p1[1];
            v16bf a;
            a[0]  = (__bf16)a0.x; a[1]  = (__bf16)a0.y; a[2]  = (__bf16)a0.z; a[3]  = (__bf16)a0.w;
            a[4]  = (__bf16)a1.x; a[5]  = (__bf16)a1.y; a[6]  = (__bf16)a1.z; a[7]  = (__bf16)a1.w;
            a[8]  = (__bf16)b0.x; a[9]  = (__bf16)b0.y; a[10] = (__bf16)b0.z; a[11] = (__bf16)b0.w;
            a[12] = (__bf16)b1.x; a[13] = (__bf16)b1.y; a[14] = (__bf16)b1.z; a[15] = (__bf16)b1.w;
            afrag[t][c] = a;
        }
    }

    // ---- Stage 3: sweep 32 column tiles; shared B, dual accumulator chains ----
    float best[2][8];
    int   bidx[2][8];
    #pragma unroll
    for (int t = 0; t < 2; ++t)
        #pragma unroll
        for (int i = 0; i < 8; ++i) { best[t][i] = -3.4e38f; bidx[t][i] = 0; }

    // B-layout: lane holds column N=l, K rows {8h..8h+7, 16+8h..16+8h+7} per 32-chunk
    const __bf16* bp = sB + (size_t)l * ROWPAD + 8 * h;
    const float*  hp = sHalf + l;
    int ncol = l;

    for (int nt = 0; nt < 32; ++nt) {
        v16bf bf[4];
        #pragma unroll
        for (int c = 0; c < 4; ++c) {
            const __bf16* q = bp + c * 32;
            v8bf lo = *(const v8bf*)q;        // 16B ds_read (16B aligned)
            v8bf hi = *(const v8bf*)(q + 16);
            v16bf b;
            #pragma unroll
            for (int e = 0; e < 8; ++e) { b[e] = lo[e]; b[e + 8] = hi[e]; }
            bf[c] = b;
        }
        v8f acc0 = {0, 0, 0, 0, 0, 0, 0, 0};
        v8f acc1 = {0, 0, 0, 0, 0, 0, 0, 0};
        #pragma unroll
        for (int c = 0; c < 4; ++c)
            acc0 = __builtin_amdgcn_wmma_f32_16x16x32_bf16(
                false, afrag[0][c], false, bf[c], (short)0, acc0, false, false);
        #pragma unroll
        for (int c = 0; c < 4; ++c)
            acc1 = __builtin_amdgcn_wmma_f32_16x16x32_bf16(
                false, afrag[1][c], false, bf[c], (short)0, acc1, false, false);

        const float hn = *hp;
        #pragma unroll
        for (int i = 0; i < 8; ++i) {
            float s = acc0[i] - hn;           // argmax(dot - 0.5||e||^2) == argmin dist
            if (s > best[0][i]) { best[0][i] = s; bidx[0][i] = ncol; }
        }
        #pragma unroll
        for (int i = 0; i < 8; ++i) {
            float s = acc1[i] - hn;
            if (s > best[1][i]) { best[1][i] = s; bidx[1][i] = ncol; }
        }
        bp   += 16 * ROWPAD;
        hp   += 16;
        ncol += 16;
    }

    // ---- Stage 4: argmax butterfly across the 16 lanes of each half-wave ----
    #pragma unroll
    for (int t = 0; t < 2; ++t) {
        #pragma unroll
        for (int off = 8; off >= 1; off >>= 1) {
            #pragma unroll
            for (int i = 0; i < 8; ++i) {
                float ov = __shfl_xor(best[t][i], off, 16);
                int   oi = __shfl_xor(bidx[t][i], off, 16);
                if (ov > best[t][i] || (ov == best[t][i] && oi < bidx[t][i])) {
                    best[t][i] = ov; bidx[t][i] = oi;   // first-index tie-break
                }
            }
        }
    }

    // ---- Stage 5: write indices ----
    #pragma unroll
    for (int t = 0; t < 2; ++t) {
        int myIdx = bidx[t][0];
        #pragma unroll
        for (int i = 1; i < 8; ++i) if (l == i) myIdx = bidx[t][i];
        if (l < 8) indOut[mBase + 16 * t + h * 8 + l] = (float)myIdx;
    }

    // ---- Stage 6: gather f32 code rows, write z_q, accumulate sumsq ----
    float sumsq = 0.0f;
    #pragma unroll
    for (int t = 0; t < 2; ++t) {
        #pragma unroll
        for (int r = 0; r < 8; ++r) {
            const int row = mBase + 16 * t + h * 8 + r;
            const int idx = bidx[t][r];   // uniform across the 16-lane group
            const float4* ep = (const float4*)(ew + (size_t)idx * D_SUB + 8 * l);
            const float4* zp = (const float4*)(z  + (size_t)row * D_SUB + 8 * l);
            float4 q0 = ep[0], q1 = ep[1];
            float4 f0 = zp[0], f1 = zp[1];
            float4* op = (float4*)(outq + (size_t)row * D_SUB + 8 * l);
            op[0] = q0; op[1] = q1;
            float d0 = q0.x - f0.x, d1 = q0.y - f0.y, d2 = q0.z - f0.z, d3 = q0.w - f0.w;
            float d4 = q1.x - f1.x, d5 = q1.y - f1.y, d6 = q1.z - f1.z, d7 = q1.w - f1.w;
            sumsq += d0 * d0 + d1 * d1 + d2 * d2 + d3 * d3
                   + d4 * d4 + d5 * d5 + d6 * d6 + d7 * d7;
        }
    }

    // ---- Stage 7: loss reduction: wave shuffle -> LDS -> one atomic per WG ----
    #pragma unroll
    for (int off = 16; off >= 1; off >>= 1) sumsq += __shfl_xor(sumsq, off, 32);
    if (lane == 0) sRed[wave] = sumsq;
    __syncthreads();
    if (tid == 0) {
        float s = 0.0f;
        #pragma unroll
        for (int i = 0; i < 8; ++i) s += sRed[i];
        atomicAdd(diffSlot, s * DIFF_SCALE);
    }
}

extern "C" void kernel_launch(void* const* d_in, const int* in_sizes, int n_in,
                              void* d_out, int out_size, void* d_ws, size_t ws_size,
                              hipStream_t stream) {
    const float* z  = (const float*)d_in[0];   // [8,4096,512] f32
    const float* ew = (const float*)d_in[1];   // [512,128]    f32
    float* out      = (float*)d_out;
    // d_out layout (tuple concat): z_q_st [16777216] | diff [1] | ind [131072]
    float* outq     = out;
    float* diffSlot = out + 16777216;
    float* indOut   = out + 16777217;

    vq_zero_kernel<<<1, 1, 0, stream>>>(diffSlot);

    const size_t shmem = (size_t)NCODE * ROWPAD * 2   // bf16 codebook
                       + (size_t)NCODE * 4            // half norms
                       + 8 * 4;                       // per-wave partials
    vq_main_kernel<<<M_TOTAL / ROWS_PER_WG, 256, shmem, stream>>>(
        z, ew, outq, diffSlot, indOut);
}
